// MultiHeadAttentionSequence_48610439856289
// MI455X (gfx1250) — matrix operand order repaired and checked by reading, not compiled, not run on hardware
//
#include <hip/hip_runtime.h>

typedef __attribute__((ext_vector_type(16))) _Float16 v16h;
typedef __attribute__((ext_vector_type(8)))  float    v8f;
typedef __attribute__((ext_vector_type(4)))  unsigned int u32x4;
typedef __attribute__((ext_vector_type(8)))  int      i32x8;
typedef __attribute__((ext_vector_type(4)))  int      i32x4;

#define B_  4
#define S_  2048
#define D_  1024
#define H_  16
#define DK_ 64
#define DV_ 64
#define M_  (B_ * S_)   // 8192 rows of flattened [B*S, D]

// ---------------------------------------------------------------------------
// Fragment loaders (wave32, 16x16x32 f16 WMMA)
// ---------------------------------------------------------------------------
__device__ __forceinline__ v16h load16h(const _Float16* p0, const _Float16* p1) {
  union { v16h h; float4 f[2]; } u;
  u.f[0] = *(const float4*)p0;   // 8 halves
  u.f[1] = *(const float4*)p1;   // 8 halves
  return u.h;
}

// A-matrix 16x32 f16, row-major source with leading dim lda (in halves).
__device__ __forceinline__ v16h load_a16(const _Float16* A, int lda, int row0,
                                         int k0, int lane) {
  const int row  = row0 + (lane & 15);
  const int koff = k0 + ((lane >> 4) << 3);
  const _Float16* p = A + (size_t)row * lda + koff;
  return load16h(p, p + 16);
}

// B-matrix 32x16 f16 from a column-major source Bt[col*ldb + k].
__device__ __forceinline__ v16h load_b16t(const _Float16* Bt, int ldb, int n0,
                                          int k0, int lane) {
  const _Float16* p = Bt + (size_t)(n0 + (lane & 15)) * ldb + k0 + ((lane >> 4) << 3);
  return load16h(p, p + 16);
}

__device__ __forceinline__ v8f wmma_f16(v16h a, v16h b, v8f c) {
  return __builtin_amdgcn_wmma_f32_16x16x32_f16(false, a, false, b, (short)0, c,
                                                false, false);
}

// ---------------------------------------------------------------------------
// Tensor Data Mover: 2D tile load (2-byte elements) global -> LDS.
// Descriptor per CDNA5 ISA ch.8 (group0/group1 bitfields). Groups 2/3 unused.
// ---------------------------------------------------------------------------
__device__ __forceinline__ void tdm_load_2d(unsigned lds_addr, const void* gptr,
                                            unsigned tile_w, unsigned tile_h,
                                            unsigned long long row_stride) {
  const unsigned long long ga = (unsigned long long)(size_t)gptr;
  u32x4 g0;
  g0[0] = 1u;                                           // count=1, user desc
  g0[1] = lds_addr;                                     // lds_addr [63:32]
  g0[2] = (unsigned)(ga & 0xffffffffull);               // global_addr[31:0]
  g0[3] = (unsigned)((ga >> 32) & 0x01ffffffull)        // global_addr[56:32]
          | 0x80000000u;                                // type=2 ("image")
  i32x8 g1;
  g1[0] = 0x00010000;                                   // data_size=1 (2 bytes)
  g1[1] = (int)((tile_w & 0xffffu) << 16);              // tensor_dim0 lo
  g1[2] = (int)(((tile_w >> 16) & 0xffffu) |            // tensor_dim0 hi
                ((tile_h & 0xffffu) << 16));            // tensor_dim1 lo
  g1[3] = (int)(((tile_h >> 16) & 0xffffu) |            // tensor_dim1 hi
                ((tile_w & 0xffffu) << 16));            // tile_dim0
  g1[4] = (int)(tile_h & 0xffffu);                      // tile_dim1 (dim2=0)
  g1[5] = (int)(row_stride & 0xffffffffull);            // dim0_stride lo
  g1[6] = (int)((row_stride >> 32) & 0xffffull);        // dim0_stride hi
  g1[7] = 0;
  i32x4 gz = {0, 0, 0, 0};
#if defined(__clang_major__) && (__clang_major__ >= 23)
  i32x8 gz8 = {};
  __builtin_amdgcn_tensor_load_to_lds(g0, g1, gz, gz, gz8, 0);
#else
  __builtin_amdgcn_tensor_load_to_lds(g0, g1, gz, gz, 0);
#endif
}

// ---------------------------------------------------------------------------
// Prep kernels
// ---------------------------------------------------------------------------
__global__ void cvt_f32_to_f16_kernel(const float* __restrict__ src,
                                      _Float16* __restrict__ dst, int n) {
  int i = blockIdx.x * blockDim.x + threadIdx.x;
  if (i < n) dst[i] = (_Float16)src[i];
}

__global__ void transpose_w_kernel(const float* __restrict__ W,
                                   _Float16* __restrict__ Wt) {
  int i = blockIdx.x * blockDim.x + threadIdx.x;
  int n = i >> 10;
  int d = i & 1023;
  Wt[i] = (_Float16)W[d * D_ + n];
}

// ---------------------------------------------------------------------------
// Projection GEMM: 32x64 tile per wave (8 WMMAs / k-step; 4x A-frag reuse,
// 2x B-frag reuse) -- sized to stay inside the VGPR budget without spills.
// mode 0/1: write [B,H,S,64] (Q, K). mode 2: write Vt [B,H,64,S].
// ---------------------------------------------------------------------------
__global__ void __launch_bounds__(32)
proj_gemm_kernel(const _Float16* __restrict__ X,
                 const _Float16* __restrict__ Wt,
                 const float* __restrict__ bias,
                 _Float16* __restrict__ out, int mode) {
  const int lane = threadIdx.x & 31;
  const int n0 = blockIdx.x * 64;
  const int m0 = blockIdx.y * 32;
  v8f acc[2][4] = {};
  for (int k = 0; k < D_; k += 32) {
    v16h a0 = load_a16(X, D_, m0,      k, lane);
    v16h a1 = load_a16(X, D_, m0 + 16, k, lane);
#pragma unroll
    for (int j = 0; j < 4; ++j) {
      v16h b = load_b16t(Wt, D_, n0 + j * 16, k, lane);
      acc[0][j] = wmma_f16(a0, b, acc[0][j]);
      acc[1][j] = wmma_f16(a1, b, acc[1][j]);
    }
  }
  const int base = (lane >> 4) << 3;
#pragma unroll
  for (int j = 0; j < 4; ++j) {
    const int col = n0 + j * 16 + (lane & 15);
    const float bn = bias[col];
    const int h  = col >> 6;
    const int dk = col & 63;
#pragma unroll
    for (int i = 0; i < 2; ++i) {
#pragma unroll
      for (int r = 0; r < 8; ++r) {
        int m  = m0 + i * 16 + base + r;
        int bb = m >> 11;
        int s  = m & (S_ - 1);
        float v = acc[i][j][r] + bn;
        size_t idx;
        if (mode == 2) idx = (((size_t)(bb * H_ + h) * DV_ + dk) * S_ + s);
        else           idx = (((size_t)(bb * H_ + h) * S_ + s) * DK_ + dk);
        out[idx] = (_Float16)v;
      }
    }
  }
}

// ---------------------------------------------------------------------------
// Attention: 8 waves per block, 16 queries per wave (128 per block).
// K/V kv-tiles staged into LDS by the Tensor Data Mover (double-buffered),
// shared by all 8 waves. Two passes: online softmax stats, then normalized
// attn write + fused P@V (P transposed C->A layout through per-wave LDS).
// ---------------------------------------------------------------------------
__global__ void __launch_bounds__(256)
attn_kernel(const _Float16* __restrict__ Qh,
            const _Float16* __restrict__ Kh,
            const _Float16* __restrict__ Vt,
            float* __restrict__ attn_out,
            _Float16* __restrict__ heads) {
  const int tid  = threadIdx.x;
  const int lane = tid & 31;
  const int wave = tid >> 5;
  const int q0 = blockIdx.x * 128 + wave * 16;
  const int h  = blockIdx.y;
  const int b  = blockIdx.z;
  const size_t bh = (size_t)b * H_ + h;

  const _Float16* Qbh = Qh + bh * (size_t)S_ * DK_;   // [S, 64]
  const _Float16* Kbh = Kh + bh * (size_t)S_ * DK_;   // [S, 64]
  const _Float16* Vbh = Vt + bh * (size_t)DV_ * S_;   // [64, S]
  float* attn_bh = attn_out + bh * (size_t)S_ * S_;

  const float scale = 0.125f;  // 1/sqrt(64)

  __shared__ __align__(16) _Float16 Kt[2][32 * 64];   // kv-major [32][64]
  __shared__ __align__(16) _Float16 Vl[2][64 * 32];   // dv-major [64][32]
  __shared__ __align__(16) _Float16 pT[8][16 * 32];   // per-wave P tile

  const unsigned kAddr[2] = {(unsigned)(size_t)&Kt[0][0],
                             (unsigned)(size_t)&Kt[1][0]};
  const unsigned vAddr[2] = {(unsigned)(size_t)&Vl[0][0],
                             (unsigned)(size_t)&Vl[1][0]};

  // Q fragments for this wave's 16 rows (K = 64 -> two 32-wide steps)
  const v16h aq0 = load_a16(Qbh, DK_, q0, 0,  lane);
  const v16h aq1 = load_a16(Qbh, DK_, q0, 32, lane);

  // ---- pass 1: online softmax statistics (K tiles via TDM) ----
  float mrow[8], lrow[8];
#pragma unroll
  for (int r = 0; r < 8; ++r) { mrow[r] = -3.0e38f; lrow[r] = 0.0f; }

  if (wave == 0) tdm_load_2d(kAddr[0], Kbh, 64, 32, DK_);
  for (int kc = 0; kc < S_; kc += 32) {
    const int cur = (kc >> 5) & 1;
    if (wave == 0) __builtin_amdgcn_s_wait_tensorcnt(0);
    __syncthreads();
    if (wave == 0 && kc + 32 < S_)
      tdm_load_2d(kAddr[1 - cur], Kbh + (size_t)(kc + 32) * DK_, 64, 32, DK_);
#pragma unroll
    for (int t = 0; t < 2; ++t) {
      v8f sc = {};
      sc = wmma_f16(aq0, load_b16t(Kt[cur], 64, t * 16, 0,  lane), sc);
      sc = wmma_f16(aq1, load_b16t(Kt[cur], 64, t * 16, 32, lane), sc);
#pragma unroll
      for (int r = 0; r < 8; ++r) {
        float s  = sc[r] * scale;
        float mo = mrow[r];
        float mn = fmaxf(mo, s);
        lrow[r] = lrow[r] * __expf(mo - mn) + __expf(s - mn);
        mrow[r] = mn;
      }
    }
  }
  // merge statistics across the 16 lanes of each half (same rows per half)
#pragma unroll
  for (int mask = 1; mask < 16; mask <<= 1) {
#pragma unroll
    for (int r = 0; r < 8; ++r) {
      float mo = __shfl_xor(mrow[r], mask, 32);
      float lo = __shfl_xor(lrow[r], mask, 32);
      float mn = fmaxf(mrow[r], mo);
      lrow[r] = lrow[r] * __expf(mrow[r] - mn) + lo * __expf(mo - mn);
      mrow[r] = mn;
    }
  }
  float invl[8];
#pragma unroll
  for (int r = 0; r < 8; ++r) invl[r] = 1.0f / lrow[r];

  // ---- pass 2: normalized P -> attn output, fused P@V ----
  v8f oacc[4] = {{}, {}, {}, {}};
  const int base = (lane >> 4) << 3;
  const int col  = lane & 15;

  __syncthreads();
  if (wave == 0) {
    tdm_load_2d(kAddr[0], Kbh, 64, 32, DK_);
    tdm_load_2d(vAddr[0], Vbh, 32, 64, S_);
  }
  for (int kc = 0; kc < S_; kc += 32) {
    const int cur = (kc >> 5) & 1;
    if (wave == 0) __builtin_amdgcn_s_wait_tensorcnt(0);
    __syncthreads();
    if (wave == 0 && kc + 32 < S_) {
      tdm_load_2d(kAddr[1 - cur], Kbh + (size_t)(kc + 32) * DK_, 64, 32, DK_);
      tdm_load_2d(vAddr[1 - cur], Vbh + (kc + 32), 32, 64, S_);
    }
#pragma unroll
    for (int t = 0; t < 2; ++t) {
      v8f sc = {};
      sc = wmma_f16(aq0, load_b16t(Kt[cur], 64, t * 16, 0,  lane), sc);
      sc = wmma_f16(aq1, load_b16t(Kt[cur], 64, t * 16, 32, lane), sc);
#pragma unroll
      for (int r = 0; r < 8; ++r) {
        float p = __expf(sc[r] * scale - mrow[r]) * invl[r];
        attn_bh[(size_t)(q0 + base + r) * S_ + kc + t * 16 + col] = p;
        pT[wave][(base + r) * 32 + t * 16 + col] = (_Float16)p;
      }
    }
    // per-wave LDS transpose C-layout -> A-layout (DS ops in-order per wave)
    const v16h ap = load_a16(pT[wave], 32, 0, 0, lane);
#pragma unroll
    for (int nt = 0; nt < 4; ++nt)
      oacc[nt] = wmma_f16(ap, load_b16t(Vl[cur], 32, nt * 16, 0, lane), oacc[nt]);
  }

  // write concatenated-heads activations [B*S, H*DV] in f16
#pragma unroll
  for (int nt = 0; nt < 4; ++nt)
#pragma unroll
    for (int r = 0; r < 8; ++r) {
      int m = b * S_ + q0 + base + r;
      heads[(size_t)m * (H_ * DV_) + h * DV_ + nt * 16 + col] =
          (_Float16)oacc[nt][r];
    }
}

// ---------------------------------------------------------------------------
// Output projection: Y[M,1024] = heads @ Wo + bo, 32x64 tile per wave
// ---------------------------------------------------------------------------
__global__ void __launch_bounds__(32)
out_gemm_kernel(const _Float16* __restrict__ heads,
                const _Float16* __restrict__ Wot,
                const float* __restrict__ bo,
                float* __restrict__ Y) {
  const int lane = threadIdx.x & 31;
  const int n0 = blockIdx.x * 64;
  const int m0 = blockIdx.y * 32;
  v8f acc[2][4] = {};
  for (int k = 0; k < D_; k += 32) {
    v16h a0 = load_a16(heads, D_, m0,      k, lane);
    v16h a1 = load_a16(heads, D_, m0 + 16, k, lane);
#pragma unroll
    for (int j = 0; j < 4; ++j) {
      v16h b = load_b16t(Wot, D_, n0 + j * 16, k, lane);
      acc[0][j] = wmma_f16(a0, b, acc[0][j]);
      acc[1][j] = wmma_f16(a1, b, acc[1][j]);
    }
  }
  const int base = (lane >> 4) << 3;
#pragma unroll
  for (int j = 0; j < 4; ++j) {
    const int col = n0 + j * 16 + (lane & 15);
    const float bn = bo[col];
#pragma unroll
    for (int i = 0; i < 2; ++i)
#pragma unroll
      for (int r = 0; r < 8; ++r)
        Y[(size_t)(m0 + i * 16 + base + r) * D_ + col] = acc[i][j][r] + bn;
  }
}

// ---------------------------------------------------------------------------
// Residual + LayerNorm over rows of 1024, one 256-thread block per row
// ---------------------------------------------------------------------------
__global__ void ln_kernel(const float* __restrict__ Y,
                          const float* __restrict__ qres,
                          const float* __restrict__ gamma,
                          const float* __restrict__ beta,
                          float* __restrict__ out) {
  const int row = blockIdx.x;
  const int tid = threadIdx.x;
  const float* y = Y    + (size_t)row * D_;
  const float* x = qres + (size_t)row * D_;
  __shared__ float red[256];

  float vals[4];
  float s = 0.0f;
#pragma unroll
  for (int i = 0; i < 4; ++i) {
    vals[i] = y[tid + i * 256] + x[tid + i * 256];
    s += vals[i];
  }
  red[tid] = s;
  __syncthreads();
  for (int off = 128; off > 0; off >>= 1) {
    if (tid < off) red[tid] += red[tid + off];
    __syncthreads();
  }
  const float mu = red[0] * (1.0f / D_);
  __syncthreads();

  float vs = 0.0f;
#pragma unroll
  for (int i = 0; i < 4; ++i) {
    float d = vals[i] - mu;
    vs += d * d;
  }
  red[tid] = vs;
  __syncthreads();
  for (int off = 128; off > 0; off >>= 1) {
    if (tid < off) red[tid] += red[tid + off];
    __syncthreads();
  }
  const float rstd = rsqrtf(red[0] * (1.0f / D_) + 1e-5f);
#pragma unroll
  for (int i = 0; i < 4; ++i) {
    int j = tid + i * 256;
    out[(size_t)row * D_ + j] = (vals[i] - mu) * rstd * gamma[j] + beta[j];
  }
}

// ---------------------------------------------------------------------------
// Host-side launch
// ---------------------------------------------------------------------------
extern "C" void kernel_launch(void* const* d_in, const int* in_sizes, int n_in,
                              void* d_out, int out_size, void* d_ws, size_t ws_size,
                              hipStream_t stream) {
  const float* q     = (const float*)d_in[0];
  const float* k     = (const float*)d_in[1];
  const float* v     = (const float*)d_in[2];
  const float* Wq    = (const float*)d_in[3];
  const float* bq    = (const float*)d_in[4];
  const float* Wk    = (const float*)d_in[5];
  const float* bk    = (const float*)d_in[6];
  const float* Wv    = (const float*)d_in[7];
  const float* bv    = (const float*)d_in[8];
  const float* Wo    = (const float*)d_in[9];
  const float* bo    = (const float*)d_in[10];
  const float* gamma = (const float*)d_in[11];
  const float* beta  = (const float*)d_in[12];

  float* out  = (float*)d_out;                       // [B,S,D]
  float* attn = out + (size_t)B_ * S_ * D_;          // [B,H,S,S]

  char* w = (char*)d_ws;
  const size_t szAct = (size_t)M_ * D_ * sizeof(_Float16);  // 16 MB
  const size_t szW   = (size_t)D_ * D_ * sizeof(_Float16);  //  2 MB
  _Float16* qh    = (_Float16*)(w);                 w += szAct;
  _Float16* kh    = (_Float16*)(w);                 w += szAct;
  _Float16* vh    = (_Float16*)(w);                 w += szAct;
  _Float16* Wqt   = (_Float16*)(w);                 w += szW;
  _Float16* Wkt   = (_Float16*)(w);                 w += szW;
  _Float16* Wvt   = (_Float16*)(w);                 w += szW;
  _Float16* Wot   = (_Float16*)(w);                 w += szW;
  _Float16* Qh    = (_Float16*)(w);                 w += szAct; // [B,H,S,64]
  _Float16* Kh    = (_Float16*)(w);                 w += szAct; // [B,H,S,64]
  _Float16* Vth   = (_Float16*)(w);                 w += szAct; // [B,H,64,S]
  _Float16* heads = (_Float16*)(w);                 w += szAct; // [B*S, 1024]
  float*    Y     = (float*)(w);                    // [B*S, 1024] f32

  const int nAct = M_ * D_;
  const int cvtBlocks = (nAct + 255) / 256;
  cvt_f32_to_f16_kernel<<<cvtBlocks, 256, 0, stream>>>(q, qh, nAct);
  cvt_f32_to_f16_kernel<<<cvtBlocks, 256, 0, stream>>>(k, kh, nAct);
  cvt_f32_to_f16_kernel<<<cvtBlocks, 256, 0, stream>>>(v, vh, nAct);

  const int trBlocks = (D_ * D_) / 256;
  transpose_w_kernel<<<trBlocks, 256, 0, stream>>>(Wq, Wqt);
  transpose_w_kernel<<<trBlocks, 256, 0, stream>>>(Wk, Wkt);
  transpose_w_kernel<<<trBlocks, 256, 0, stream>>>(Wv, Wvt);
  transpose_w_kernel<<<trBlocks, 256, 0, stream>>>(Wo, Wot);

  dim3 gGemm(D_ / 64, M_ / 32);                      // (16, 256)
  proj_gemm_kernel<<<gGemm, 32, 0, stream>>>(qh, Wqt, bq, Qh, 0);
  proj_gemm_kernel<<<gGemm, 32, 0, stream>>>(kh, Wkt, bk, Kh, 1);
  proj_gemm_kernel<<<gGemm, 32, 0, stream>>>(vh, Wvt, bv, Vth, 2);

  dim3 gAttn(S_ / 128, H_, B_);                      // (16, 16, 4)
  attn_kernel<<<gAttn, 256, 0, stream>>>(Qh, Kh, Vth, attn, heads);

  out_gemm_kernel<<<gGemm, 32, 0, stream>>>(heads, Wot, bo, Y);

  ln_kernel<<<M_, 256, 0, stream>>>(Y, q, gamma, beta, out);
}
